// LSTM_5171140625025
// MI455X (gfx1250) — matrix operand order repaired and checked by reading, not compiled, not run on hardware
//
#include <hip/hip_runtime.h>
#include <hip/hip_bf16.h>

// ---------------------------------------------------------------------------
// CDNA5 (gfx1250) WMMA types & helpers
// ---------------------------------------------------------------------------
typedef __attribute__((ext_vector_type(16))) _Float16 v16h;
typedef __attribute__((ext_vector_type(8)))  float    v8f;
typedef unsigned int v4u __attribute__((ext_vector_type(4)));

union Frag16 { v4u u[2]; v16h h; };   // 32B: one WMMA A/B fragment per lane

#define WMMA_F16(a, b, c) \
    __builtin_amdgcn_wmma_f32_16x16x32_f16(false, (a), false, (b), (short)0, (c), false, false)

static __device__ __forceinline__ float sigmoidf_(float x) {
    return 1.0f / (1.0f + __expf(-x));
}

// ---------------------------------------------------------------------------
// Kernel 1: embedding GEMM, K-split for occupancy.
//   EF[5888,304] += X[5888,8150(seg)] * W[8150(seg),300]   (f32 atomics)
// grid = (46 M-blocks, 4 K-segments, 2 channels), 256 threads / WG.
// 8 waves; wave w owns rows [m0+16w, m0+16w+16), all 19 col tiles.
// Staging uses fixed per-thread coordinates + constant strides so all
// addressing folds into instruction offsets (keeps VGPRs < 256 ->
// 2 waves/SIMD to hide the DS->WMMA waits).
// ---------------------------------------------------------------------------
#define EMB_K   8150
#define EMB_KT  255      // ceil(8150/32) K-chunks total
#define EMB_SEG 64       // K-chunks per segment (4 segments: 64,64,64,63)
#define EMB_NT  19       // 304 output columns
#define EF_LD   304      // f32 partial-accumulator row stride
#define E_LD    320      // f16 row stride of E (zero-padded K for next GEMM)

__global__ __launch_bounds__(256) __attribute__((amdgpu_waves_per_eu(2)))
void embed_gemm_kernel(const float* __restrict__ X1, const float* __restrict__ X2,
                       const float* __restrict__ W1, const float* __restrict__ W2,
                       float* __restrict__ EF1, float* __restrict__ EF2)
{
    const int ch = blockIdx.z;
    const float* __restrict__ X = ch ? X2 : X1;
    const float* __restrict__ W = ch ? W2 : W1;
    float* __restrict__ EF = ch ? EF2 : EF1;

    __shared__ _Float16 Ab[128][40];    // 128 rows x 32 K (+8 pad, 16B-aligned rows)
    __shared__ _Float16 Bb[304][40];    // 304 cols x 32 K, transposed

    const int tid   = threadIdx.x;
    const int wv    = tid >> 5;
    const int lane  = tid & 31;
    const int lmod  = lane & 15;
    const int lhalf = lane >> 4;
    const int m0    = blockIdx.x * 128;

    const int cb0 = blockIdx.y * EMB_SEG;
    const int cb1 = min(cb0 + EMB_SEG, EMB_KT);

    // fixed per-thread staging coordinates
    const int p  = tid & 15;    // A: K-pair within chunk (k = 2p)
    const int r0 = tid >> 4;    // A: base row; r = r0 + 16*s
    const int bk = tid >> 3;    // B: k within chunk (0..31)
    const int bq = tid & 7;     // B: base col pair; n = 2*bq + 16*s

    const float* xbase = X + (size_t)(m0 + r0) * EMB_K + p * 2;
    const float* wbase = W + (size_t)bk * 300 + bq * 2;
    _Float16* abase = &Ab[r0][p * 2];
    _Float16* bbase = &Bb[bq * 2][bk];

    v8f acc[EMB_NT];
    v8f vzero = {};
    #pragma unroll
    for (int j = 0; j < EMB_NT; ++j) acc[j] = vzero;

    for (int kb = cb0; kb < cb1; ++kb) {
        const int k0 = kb * 32;
        const float* xk = xbase + k0;
        const float* wk = wbase + (size_t)k0 * 300;
        const bool aval = (k0 + p * 2) < EMB_K;   // pairs never straddle (K even)
        const bool bval = (k0 + bk) < EMB_K;

        // ---- stage A tile (f32 -> f16): row r0+16s, constant strides ----
        #pragma unroll
        for (int s = 0; s < 8; ++s) {
            float2 v = make_float2(0.f, 0.f);
            if (aval) v = *(const float2*)(xk + (size_t)s * 16 * EMB_K);
            abase[s * 16 * 40]     = (_Float16)v.x;
            abase[s * 16 * 40 + 1] = (_Float16)v.y;
            if (p == 0 && kb + 1 < cb1)
                __builtin_prefetch(xk + (size_t)s * 16 * EMB_K + 32, 0, 1);
        }
        // ---- stage B tile transposed (f32 -> f16): n = 2bq+16s ----
        #pragma unroll
        for (int s = 0; s < 19; ++s) {
            int n = 2 * bq + 16 * s;
            float2 v = make_float2(0.f, 0.f);
            if (bval && n < 299) v = *(const float2*)(wk + s * 16);
            bbase[s * 640]      = (_Float16)v.x;   // Bb[n][bk]
            bbase[s * 640 + 40] = (_Float16)v.y;   // Bb[n+1][bk]
        }
        __syncthreads();

        Frag16 a;
        a.u[0] = *(const v4u*)&Ab[wv * 16 + lmod][lhalf * 8];
        a.u[1] = *(const v4u*)&Ab[wv * 16 + lmod][16 + lhalf * 8];
        #pragma unroll
        for (int j = 0; j < EMB_NT; ++j) {
            Frag16 b;
            b.u[0] = *(const v4u*)&Bb[j * 16 + lmod][lhalf * 8];
            b.u[1] = *(const v4u*)&Bb[j * 16 + lmod][16 + lhalf * 8];
            acc[j] = WMMA_F16(a.h, b.h, acc[j]);
        }
        __syncthreads();
    }

    // accumulate K-segment partials into f32 buffer
    #pragma unroll
    for (int j = 0; j < EMB_NT; ++j) {
        int col = j * 16 + lmod;
        #pragma unroll
        for (int r = 0; r < 8; ++r) {
            int row = m0 + wv * 16 + (lhalf << 3) + r;
            atomicAdd(&EF[(size_t)row * EF_LD + col], acc[j][r]);
        }
    }
}

// ---------------------------------------------------------------------------
// Kernel 1b: EF f32 [5888,304] -> E f16 [5888,320] (zero-pad cols 304..319)
// ---------------------------------------------------------------------------
__global__ __launch_bounds__(256)
void e_convert_kernel(const float* __restrict__ EF, _Float16* __restrict__ E16)
{
    int idx = blockIdx.x * 256 + threadIdx.x;
    if (idx >= 5888 * E_LD) return;
    int row = idx / E_LD, col = idx % E_LD;
    float v = (col < EF_LD) ? EF[(size_t)row * EF_LD + col] : 0.f;
    E16[idx] = (_Float16)v;
}

// ---------------------------------------------------------------------------
// Kernel 2: generic per-wave-tile GEMM, A[M,lda]f16 row-major, BT[N,ldb]f16
// (B pre-transposed). One 16x16 tile per wave, fragments from global (L2),
// software-pipelined one K-step ahead. flags: bit0 = relu, bit1 = f16 output
// ---------------------------------------------------------------------------
__global__ __launch_bounds__(256)
void tile_gemm_f16(const _Float16* __restrict__ A, int lda,
                   const _Float16* __restrict__ BT, int ldb,
                   const float* __restrict__ bias,
                   void* __restrict__ Cout, int ldc,
                   int mtiles, int ntiles, int ksteps, int flags)
{
    const int tid = threadIdx.x;
    const int wv = tid >> 5;
    const int lane = tid & 31, lmod = lane & 15, lhalf = lane >> 4;
    int tile = blockIdx.x * 8 + wv;
    if (tile >= mtiles * ntiles) return;
    int mt = tile / ntiles, nt = tile % ntiles;

    v8f acc = {};
    const _Float16* arow = A + (size_t)(mt * 16 + lmod) * lda;
    const _Float16* brow = BT + (size_t)(nt * 16 + lmod) * ldb;

    Frag16 a, b;
    a.u[0] = *(const v4u*)(arow + lhalf * 8);
    a.u[1] = *(const v4u*)(arow + 16 + lhalf * 8);
    b.u[0] = *(const v4u*)(brow + lhalf * 8);
    b.u[1] = *(const v4u*)(brow + 16 + lhalf * 8);

    for (int kb = 0; kb < ksteps; ++kb) {
        Frag16 an = a, bn = b;
        if (kb + 1 < ksteps) {
            const _Float16* ap = arow + (kb + 1) * 32;
            an.u[0] = *(const v4u*)(ap + lhalf * 8);
            an.u[1] = *(const v4u*)(ap + 16 + lhalf * 8);
            const _Float16* bp = brow + (kb + 1) * 32 + lhalf * 8;
            bn.u[0] = *(const v4u*)bp;
            bn.u[1] = *(const v4u*)(bp + 16);
        }
        acc = WMMA_F16(a.h, b.h, acc);
        a = an; b = bn;
    }
    int col = nt * 16 + lmod;
    float bv = bias ? bias[col] : 0.f;
    #pragma unroll
    for (int r = 0; r < 8; ++r) {
        int row = mt * 16 + (lhalf << 3) + r;
        float v = acc[r] + bv;
        if (flags & 1) v = fmaxf(v, 0.f);
        if (flags & 2) ((_Float16*)Cout)[(size_t)row * ldc + col] = (_Float16)v;
        else           ((float*)Cout)[(size_t)row * ldc + col] = v;
    }
}

// ---------------------------------------------------------------------------
// Kernel 3: persistent LayerNorm-LSTM over T=46 steps. One WG per channel,
// 1024 threads = 32 waves (8/SIMD hides DS/VMEM latency); wave (mt,g) owns
// batch-tile mt of gate g. h kept f16 in LDS; c in registers; LN stats via
// shfl_xor butterflies.
// ---------------------------------------------------------------------------
#define T_STEPS 46

__global__ __launch_bounds__(1024)
void lstm_steps_kernel(const float* __restrict__ Zx1, const float* __restrict__ Zx2,
                       const _Float16* __restrict__ WhT1, const _Float16* __restrict__ WhT2,
                       const float* __restrict__ lng1, const float* __restrict__ lnb1,
                       const float* __restrict__ lng2, const float* __restrict__ lnb2,
                       float* __restrict__ H1, float* __restrict__ H2)
{
    const int chn = blockIdx.x;
    const float* __restrict__ Zx  = chn ? Zx2 : Zx1;
    const _Float16* __restrict__ WhT = chn ? WhT2 : WhT1;
    const float* __restrict__ lng = chn ? lng2 : lng1;
    const float* __restrict__ lnb = chn ? lnb2 : lnb1;
    float* __restrict__ Hout = chn ? H2 : H1;

    __shared__ _Float16 hS[128][136];        // recurrent h (f16), padded rows
    __shared__ _Float16 actS[3][128][136];   // sig(i), relu(j), sig(f+1)

    const int tid = threadIdx.x;
    const int wv = tid >> 5;
    const int lane = tid & 31, lmod = lane & 15, lhalf = lane >> 4;
    const int mt = wv & 7;       // batch tile
    const int g  = wv >> 3;      // gate: 0=i 1=j 2=f 3=o

    for (int idx = tid; idx < 128 * 136; idx += 1024)
        (&hS[0][0])[idx] = (_Float16)0.f;

    const int cbt = tid >> 3;           // cell-state batch for this thread
    const int hb  = (tid & 7) * 16;     // cell-state h base
    float cr[16];
    #pragma unroll
    for (int u = 0; u < 16; ++u) cr[u] = 0.f;

    float gw[8], bw[8];
    #pragma unroll
    for (int j = 0; j < 8; ++j) {
        int h = j * 16 + lmod;
        gw[j] = lng[g * 128 + h];
        bw[j] = lnb[g * 128 + h];
    }
    __syncthreads();

    for (int t = 0; t < T_STEPS; ++t) {
        // ---- recurrent GEMM: acc = h @ Wh   (M=128,N=512,K=128) ----
        v8f acc[8];
        v8f vzero = {};
        #pragma unroll
        for (int j = 0; j < 8; ++j) acc[j] = vzero;
        #pragma unroll
        for (int kb = 0; kb < 4; ++kb) {
            Frag16 a;
            a.u[0] = *(const v4u*)&hS[mt * 16 + lmod][kb * 32 + lhalf * 8];
            a.u[1] = *(const v4u*)&hS[mt * 16 + lmod][kb * 32 + 16 + lhalf * 8];
            #pragma unroll
            for (int j = 0; j < 8; ++j) {
                Frag16 b;
                const _Float16* bp =
                    WhT + (size_t)(g * 128 + j * 16 + lmod) * 128 + kb * 32 + lhalf * 8;
                b.u[0] = *(const v4u*)bp;
                b.u[1] = *(const v4u*)(bp + 16);
                acc[j] = WMMA_F16(a.h, b.h, acc[j]);
            }
        }
        __syncthreads();   // all waves done reading hS

        // ---- z = acc + Zx[t]; per-(b,gate) LayerNorm; activation ----
        #pragma unroll
        for (int r = 0; r < 8; ++r) {
            int b = mt * 16 + (lhalf << 3) + r;
            const float* zxp = Zx + ((size_t)b * T_STEPS + t) * 512 + g * 128 + lmod;
            float zv[8];
            float s = 0.f;
            #pragma unroll
            for (int j = 0; j < 8; ++j) { zv[j] = acc[j][r] + zxp[j * 16]; s += zv[j]; }
            s += __shfl_xor(s, 1, 32); s += __shfl_xor(s, 2, 32);
            s += __shfl_xor(s, 4, 32); s += __shfl_xor(s, 8, 32);
            float mu = s * (1.f / 128.f);
            float d2 = 0.f;
            #pragma unroll
            for (int j = 0; j < 8; ++j) { float d = zv[j] - mu; d2 += d * d; }
            d2 += __shfl_xor(d2, 1, 32); d2 += __shfl_xor(d2, 2, 32);
            d2 += __shfl_xor(d2, 4, 32); d2 += __shfl_xor(d2, 8, 32);
            float rs = rsqrtf(d2 * (1.f / 128.f) + 1e-12f);
            #pragma unroll
            for (int j = 0; j < 8; ++j) {
                int h = j * 16 + lmod;
                float nz = (zv[j] - mu) * rs * gw[j] + bw[j];
                float a;
                if (g == 0)      a = sigmoidf_(nz);
                else if (g == 1) a = fmaxf(nz, 0.f);
                else if (g == 2) a = sigmoidf_(nz + 1.f);   // forget_bias = 1
                else             a = sigmoidf_(nz);
                if (g < 3) actS[g][b][h] = (_Float16)a;
                else       hS[b][h] = (_Float16)a;          // stage sig(o) into hS
            }
        }
        __syncthreads();

        // ---- c update, cell LN, h = relu(LN(c)) * sig(o) ----
        {
            float s = 0.f, s2 = 0.f;
            #pragma unroll
            for (int u = 0; u < 16; ++u) {
                int h = hb + u;
                float ig = (float)actS[0][cbt][h];
                float jg = (float)actS[1][cbt][h];
                float fg = (float)actS[2][cbt][h];
                float cn = cr[u] * fg + ig * jg;
                cr[u] = cn; s += cn; s2 += cn * cn;
            }
            s  += __shfl_xor(s, 1, 32);  s  += __shfl_xor(s, 2, 32);  s  += __shfl_xor(s, 4, 32);
            s2 += __shfl_xor(s2, 1, 32); s2 += __shfl_xor(s2, 2, 32); s2 += __shfl_xor(s2, 4, 32);
            float mu = s * (1.f / 128.f);
            float var = s2 * (1.f / 128.f) - mu * mu;
            float rs = rsqrtf(var + 1e-12f);
            #pragma unroll
            for (int u = 0; u < 16; ++u) {
                int h = hb + u;
                float ln = (cr[u] - mu) * rs * lng[4 * 128 + h] + lnb[4 * 128 + h];
                float hv = fmaxf(ln, 0.f) * (float)hS[cbt][h];
                hS[cbt][h] = (_Float16)hv;
                Hout[((size_t)cbt * T_STEPS + t) * 128 + h] = hv;
            }
        }
        __syncthreads();
    }
}

// ---------------------------------------------------------------------------
// Kernel 4: attention scores + softmax + context (per batch, per channel)
// ---------------------------------------------------------------------------
__global__ __launch_bounds__(64)
void attn_ctx_kernel(const float* __restrict__ H1, const float* __restrict__ H2,
                     const float* __restrict__ v1, const float* __restrict__ v2,
                     _Float16* __restrict__ C1, _Float16* __restrict__ C2)
{
    const int chn = blockIdx.y;
    const float* H = chn ? H2 : H1;
    const float* v = chn ? v2 : v1;
    _Float16* C = chn ? C2 : C1;
    const int b = blockIdx.x;
    __shared__ float al[T_STEPS];
    const int tid = threadIdx.x;

    if (tid < T_STEPS) {
        const float* row = H + ((size_t)b * T_STEPS + tid) * 128;
        float s = 0.f;
        for (int h = 0; h < 128; ++h) s += row[h] * v[h];
        al[tid] = s;
    }
    __syncthreads();
    if (tid == 0) {
        float m = al[0];
        for (int t = 1; t < T_STEPS; ++t) m = fmaxf(m, al[t]);
        float sum = 0.f;
        for (int t = 0; t < T_STEPS; ++t) { float e = __expf(al[t] - m); al[t] = e; sum += e; }
        float inv = 1.f / sum;
        for (int t = 0; t < T_STEPS; ++t) al[t] *= inv;
    }
    __syncthreads();
    for (int h = tid; h < 128; h += 64) {
        float c = 0.f;
        for (int t = 0; t < T_STEPS; ++t) c += al[t] * H[((size_t)b * T_STEPS + t) * 128 + h];
        C[(size_t)b * 128 + h] = (_Float16)c;
    }
}

// ---------------------------------------------------------------------------
// Kernel 5: final logits  [128,512]f16 @ [512,2]f32 + bias
// ---------------------------------------------------------------------------
__global__ __launch_bounds__(256)
void logits_kernel(const _Float16* __restrict__ hid, const float* __restrict__ Wout,
                   const float* __restrict__ bout, float* __restrict__ out)
{
    int idx = threadIdx.x;           // 256 threads: (b, o)
    int b = idx >> 1, o = idx & 1;
    float acc = bout[o];
    for (int k = 0; k < 512; ++k)
        acc += (float)hid[(size_t)b * 512 + k] * Wout[k * 2 + o];
    out[b * 2 + o] = acc;
}

// ---------------------------------------------------------------------------
// Prep: transpose + f32->f16 convert:  dst[n][k] = src[rowoff+k][n] (zero-pad K)
// ---------------------------------------------------------------------------
__global__ __launch_bounds__(256)
void transpose_f32_f16(const float* __restrict__ src, _Float16* __restrict__ dst,
                       int N, int Kreal, int Kpad, int rowoff)
{
    int idx = blockIdx.x * 256 + threadIdx.x;
    if (idx >= N * Kpad) return;
    int n = idx / Kpad, k = idx % Kpad;
    float v = (k < Kreal) ? src[(size_t)(rowoff + k) * N + n] : 0.f;
    dst[idx] = (_Float16)v;
}

// ---------------------------------------------------------------------------
// Host orchestration
// ---------------------------------------------------------------------------
extern "C" void kernel_launch(void* const* d_in, const int* in_sizes, int n_in,
                              void* d_out, int out_size, void* d_ws, size_t ws_size,
                              hipStream_t stream)
{
    (void)in_sizes; (void)n_in; (void)out_size; (void)ws_size;
    const float* x1     = (const float*)d_in[0];
    const float* x2     = (const float*)d_in[1];
    const float* embed1 = (const float*)d_in[2];
    const float* embed2 = (const float*)d_in[3];
    const float* Wl1    = (const float*)d_in[4];
    const float* Wl2    = (const float*)d_in[5];
    const float* lng1   = (const float*)d_in[6];
    const float* lnb1   = (const float*)d_in[7];
    const float* lng2   = (const float*)d_in[8];
    const float* lnb2   = (const float*)d_in[9];
    const float* av1    = (const float*)d_in[10];
    const float* Wo1    = (const float*)d_in[11];
    const float* bo1    = (const float*)d_in[12];
    const float* av2    = (const float*)d_in[13];
    const float* Wo2    = (const float*)d_in[14];
    const float* bo2    = (const float*)d_in[15];
    const float* W_h    = (const float*)d_in[16];
    const float* b_h    = (const float*)d_in[17];
    const float* W_out  = (const float*)d_in[18];
    const float* b_out  = (const float*)d_in[19];

    char* ws = (char*)d_ws;
    size_t off = 0;
    auto alloc = [&](size_t bytes) {
        size_t o = off;
        off = (off + bytes + 255) & ~(size_t)255;
        return o;
    };
    float*    ef1  = (float*)   (ws + alloc((size_t)5888 * 304 * 4));
    float*    ef2  = (float*)   (ws + alloc((size_t)5888 * 304 * 4));
    _Float16* e1   = (_Float16*)(ws + alloc((size_t)5888 * 320 * 2));
    _Float16* e2   = (_Float16*)(ws + alloc((size_t)5888 * 320 * 2));
    float*    zx1  = (float*)   (ws + alloc((size_t)5888 * 512 * 4));
    float*    zx2  = (float*)   (ws + alloc((size_t)5888 * 512 * 4));
    _Float16* wxt1 = (_Float16*)(ws + alloc((size_t)512 * 320 * 2));
    _Float16* wxt2 = (_Float16*)(ws + alloc((size_t)512 * 320 * 2));
    _Float16* wht1 = (_Float16*)(ws + alloc((size_t)512 * 128 * 2));
    _Float16* wht2 = (_Float16*)(ws + alloc((size_t)512 * 128 * 2));
    _Float16* wot1 = (_Float16*)(ws + alloc((size_t)512 * 128 * 2));
    _Float16* wot2 = (_Float16*)(ws + alloc((size_t)512 * 128 * 2));
    _Float16* whT  = (_Float16*)(ws + alloc((size_t)512 * 1024 * 2));
    float*    h1   = (float*)   (ws + alloc((size_t)128 * 46 * 128 * 4));
    float*    h2   = (float*)   (ws + alloc((size_t)128 * 46 * 128 * 4));
    _Float16* ctx1 = (_Float16*)(ws + alloc((size_t)128 * 128 * 2));
    _Float16* ctx2 = (_Float16*)(ws + alloc((size_t)128 * 128 * 2));
    _Float16* acat = (_Float16*)(ws + alloc((size_t)128 * 1024 * 2));
    _Float16* hid  = (_Float16*)(ws + alloc((size_t)128 * 512 * 2));

    // zero the K-split partial accumulators
    hipMemsetAsync(ef1, 0, (size_t)5888 * 304 * 4, stream);
    hipMemsetAsync(ef2, 0, (size_t)5888 * 304 * 4, stream);

    // weight transposes / f16 conversion (all small, L2-resident afterwards)
    transpose_f32_f16<<<(512 * 320 + 255) / 256, 256, 0, stream>>>(Wl1, wxt1, 512, 300, 320, 0);
    transpose_f32_f16<<<(512 * 320 + 255) / 256, 256, 0, stream>>>(Wl2, wxt2, 512, 300, 320, 0);
    transpose_f32_f16<<<(512 * 128 + 255) / 256, 256, 0, stream>>>(Wl1, wht1, 512, 128, 128, 300);
    transpose_f32_f16<<<(512 * 128 + 255) / 256, 256, 0, stream>>>(Wl2, wht2, 512, 128, 128, 300);
    transpose_f32_f16<<<(512 * 128 + 255) / 256, 256, 0, stream>>>(Wo1, wot1, 512, 128, 128, 0);
    transpose_f32_f16<<<(512 * 128 + 255) / 256, 256, 0, stream>>>(Wo2, wot2, 512, 128, 128, 0);
    transpose_f32_f16<<<(512 * 1024 + 255) / 256, 256, 0, stream>>>(W_h, whT, 512, 1024, 1024, 0);

    // 1) embedding GEMMs, K-split 4 ways for machine fill (368 WGs)
    embed_gemm_kernel<<<dim3(46, 4, 2), 256, 0, stream>>>(x1, x2, embed1, embed2, ef1, ef2);
    e_convert_kernel<<<(5888 * E_LD + 255) / 256, 256, 0, stream>>>(ef1, e1);
    e_convert_kernel<<<(5888 * E_LD + 255) / 256, 256, 0, stream>>>(ef2, e2);

    // 2) hoisted input projections  Zx = e @ Wx   (M=5888,N=512,K=320)
    tile_gemm_f16<<<(368 * 32 + 7) / 8, 256, 0, stream>>>(e1, 320, wxt1, 320, nullptr,
                                                          (void*)zx1, 512, 368, 32, 10, 0);
    tile_gemm_f16<<<(368 * 32 + 7) / 8, 256, 0, stream>>>(e2, 320, wxt2, 320, nullptr,
                                                          (void*)zx2, 512, 368, 32, 10, 0);

    // 3) recurrent LayerNorm-LSTM, both channels in parallel
    lstm_steps_kernel<<<2, 1024, 0, stream>>>(zx1, zx2, wht1, wht2,
                                              lng1, lnb1, lng2, lnb2, h1, h2);

    // 4) attention contexts
    attn_ctx_kernel<<<dim3(128, 2), 64, 0, stream>>>(h1, h2, av1, av2, ctx1, ctx2);

    // 5) a1 = relu(ctx1@Wo1+bo1) -> acat[:, :512]; a2 -> acat[:, 512:]
    tile_gemm_f16<<<32, 256, 0, stream>>>(ctx1, 128, wot1, 128, bo1,
                                          (void*)acat, 1024, 8, 32, 4, 3);
    tile_gemm_f16<<<32, 256, 0, stream>>>(ctx2, 128, wot2, 128, bo2,
                                          (void*)(acat + 512), 1024, 8, 32, 4, 3);

    // 6) hidden = relu(acat @ W_h + b_h)
    tile_gemm_f16<<<32, 256, 0, stream>>>(acat, 1024, whT, 1024, b_h,
                                          (void*)hid, 512, 8, 32, 32, 3);

    // 7) logits
    logits_kernel<<<1, 256, 0, stream>>>(hid, W_out, b_out, (float*)d_out);
}
